// BiRealLinear_61529701482741
// MI455X (gfx1250) — 8-bit matrix operand order repaired and checked by reading, compiled, not run on hardware
//
#include <hip/hip_runtime.h>
#include <hip/hip_bf16.h>
#include <stdint.h>


#define M_DIM 8192
#define N_DIM 4096
#define K_DIM 4096

typedef int v8i __attribute__((ext_vector_type(8)));

__global__ __launch_bounds__(256)
void quant_x_kernel(const float* __restrict__ x, int8_t* __restrict__ xq) {
    const long base = ((long)blockIdx.x * blockDim.x + threadIdx.x) * 16;
    const float4* __restrict__ xv = (const float4*)(x + base);
    uint32_t packed[4];
#pragma unroll
    for (int j = 0; j < 4; ++j) {
        float4 v = xv[j];
        uint32_t b0 = (v.x >= 0.0f) ? 0x01u : 0xFFu;
        uint32_t b1 = (v.y >= 0.0f) ? 0x01u : 0xFFu;
        uint32_t b2 = (v.z >= 0.0f) ? 0x01u : 0xFFu;
        uint32_t b3 = (v.w >= 0.0f) ? 0x01u : 0xFFu;
        packed[j] = b0 | (b1 << 8) | (b2 << 16) | (b3 << 24);
    }
    uint4* dst = (uint4*)(xq + base);
    *dst = make_uint4(packed[0], packed[1], packed[2], packed[3]);
}

__global__ __launch_bounds__(256)
void quant_w_kernel(const float* __restrict__ w, int8_t* __restrict__ wq,
                    float* __restrict__ scale) {
    const int row = blockIdx.x;
    const int tid = threadIdx.x;
    const float* __restrict__ wrow = w + (long)row * K_DIM + tid * 16;

    float asum = 0.0f;
    uint32_t packed[4];
    const float4* __restrict__ wv = (const float4*)wrow;
#pragma unroll
    for (int j = 0; j < 4; ++j) {
        float4 v = wv[j];
        asum += fabsf(v.x) + fabsf(v.y) + fabsf(v.z) + fabsf(v.w);
        uint32_t b0 = (v.x >= 0.0f) ? 0x01u : 0xFFu;
        uint32_t b1 = (v.y >= 0.0f) ? 0x01u : 0xFFu;
        uint32_t b2 = (v.z >= 0.0f) ? 0x01u : 0xFFu;
        uint32_t b3 = (v.w >= 0.0f) ? 0x01u : 0xFFu;
        packed[j] = b0 | (b1 << 8) | (b2 << 16) | (b3 << 24);
    }
    uint4* dst = (uint4*)(wq + (long)row * K_DIM + tid * 16);
    *dst = make_uint4(packed[0], packed[1], packed[2], packed[3]);

#pragma unroll
    for (int off = 16; off > 0; off >>= 1)
        asum += __shfl_down(asum, off, 32);

    __shared__ float wave_sums[8];
    if ((tid & 31) == 0) wave_sums[tid >> 5] = asum;
    __syncthreads();
    if (tid == 0) {
        float total = 0.0f;
#pragma unroll
        for (int i = 0; i < 8; ++i) total += wave_sums[i];
        scale[row] = total * (1.0f / (float)K_DIM);
    }
}

__device__ __forceinline__ v8i load_a_frag(const int8_t* __restrict__ p) {
    int2 c0 = *(const int2*)(p);
    int2 c1 = *(const int2*)(p + 16);
    int2 c2 = *(const int2*)(p + 32);
    int2 c3 = *(const int2*)(p + 48);
    return (v8i){c0.x, c0.y, c1.x, c1.y, c2.x, c2.y, c3.x, c3.y};
}

__device__ __forceinline__ v8i load_b_frag(const int8_t* __restrict__ p) {
    int2 c0 = *(const int2*)(p);
    int2 c1 = *(const int2*)(p + 16);
    int2 c2 = *(const int2*)(p + 32);
    int2 c3 = *(const int2*)(p + 48);
    return (v8i){c0.x, c0.y, c1.x, c1.y, c2.x, c2.y, c3.x, c3.y};
}

__global__ __launch_bounds__(256)
void bireal_gemm_kernel(const int8_t* __restrict__ A,
                        const int8_t* __restrict__ B,
                        const float* __restrict__ scale,
                        float* __restrict__ out) {
    const int lane = threadIdx.x & 31;
    const int wave = threadIdx.x >> 5;
    const int wm   = wave >> 2;
    const int wn   = wave & 3;
    const int m0   = blockIdx.y * 128 + wm * 64;
    const int n0   = blockIdx.x * 256 + wn * 64;
    const int lrow = lane & 15;
    const int hi   = lane >> 4;

    int aOff[4];
    int bOff[4];
#pragma unroll
    for (int mi = 0; mi < 4; ++mi)
        aOff[mi] = (m0 + mi * 16 + lrow) * K_DIM + hi * 8;
#pragma unroll
    for (int ni = 0; ni < 4; ++ni)
        bOff[ni] = (n0 + ni * 16 + lrow) * K_DIM + hi * 8;

    v8i acc[4][4];
#pragma unroll
    for (int mi = 0; mi < 4; ++mi)
#pragma unroll
        for (int ni = 0; ni < 4; ++ni)
            acc[mi][ni] = (v8i)(0);

    v8i a0[4], b0[4], a1[4], b1[4];

#pragma unroll
    for (int mi = 0; mi < 4; ++mi) a0[mi] = load_a_frag(A + aOff[mi]);
#pragma unroll
    for (int ni = 0; ni < 4; ++ni) b0[ni] = load_b_frag(B + bOff[ni]);

    for (int k = 0; k < K_DIM; k += 128) {
        {
            const int8_t* Ak = A + (k + 64);
            const int8_t* Bk = B + (k + 64);
#pragma unroll
            for (int mi = 0; mi < 4; ++mi) a1[mi] = load_a_frag(Ak + aOff[mi]);
#pragma unroll
            for (int ni = 0; ni < 4; ++ni) b1[ni] = load_b_frag(Bk + bOff[ni]);
        }
#pragma unroll
        for (int ni = 0; ni < 4; ++ni)
#pragma unroll
            for (int mi = 0; mi < 4; ++mi)
                acc[mi][ni] = __builtin_amdgcn_wmma_i32_16x16x64_iu8(
                    true, a0[mi], true, b0[ni], acc[mi][ni], false, false);

        if (k + 128 < K_DIM) {
            const int8_t* Ak = A + (k + 128);
            const int8_t* Bk = B + (k + 128);
#pragma unroll
            for (int mi = 0; mi < 4; ++mi) a0[mi] = load_a_frag(Ak + aOff[mi]);
#pragma unroll
            for (int ni = 0; ni < 4; ++ni) b0[ni] = load_b_frag(Bk + bOff[ni]);
        }
#pragma unroll
        for (int ni = 0; ni < 4; ++ni)
#pragma unroll
            for (int mi = 0; mi < 4; ++mi)
                acc[mi][ni] = __builtin_amdgcn_wmma_i32_16x16x64_iu8(
                    true, a1[mi], true, b1[ni], acc[mi][ni], false, false);
    }

#pragma unroll
    for (int ni = 0; ni < 4; ++ni) {
        const int n = n0 + ni * 16 + lrow;
        const float s = scale[n];
#pragma unroll
        for (int mi = 0; mi < 4; ++mi) {
            const int mbase = m0 + mi * 16 + hi * 8;
#pragma unroll
            for (int v = 0; v < 8; ++v) {
                out[(long)(mbase + v) * N_DIM + n] = (float)acc[mi][ni][v] * s;
            }
        }
    }
}

extern "C" void kernel_launch(void* const* d_in, const int* in_sizes, int n_in,
                              void* d_out, int out_size, void* d_ws, size_t ws_size,
                              hipStream_t stream) {
    const float* x = (const float*)d_in[0];
    const float* w = (const float*)d_in[1];
    float* out = (float*)d_out;

    int8_t* xq    = (int8_t*)d_ws;
    int8_t* wq    = (int8_t*)d_ws + (size_t)M_DIM * K_DIM;
    float*  scale = (float*)((int8_t*)d_ws + (size_t)M_DIM * K_DIM
                                           + (size_t)N_DIM * K_DIM);

    {
        const long total = (long)M_DIM * K_DIM;
        const int blocks = (int)(total / (256 * 16));
        quant_x_kernel<<<blocks, 256, 0, stream>>>(x, xq);
    }
    quant_w_kernel<<<N_DIM, 256, 0, stream>>>(w, wq, scale);

    {
        dim3 grid(N_DIM / 256, M_DIM / 128);
        bireal_gemm_kernel<<<grid, 256, 0, stream>>>(xq, wq, scale, out);
    }
}
